// MultiHeadAttentionRoPE_15796889715401
// MI455X (gfx1250) — matrix operand order repaired and checked by reading, not compiled, not run on hardware
//
#include <hip/hip_runtime.h>

#define SEQ   4096
#define DIM   768
#define HEADS 12
#define HD    64
#define QKVN  2304   // 3*DIM

typedef __bf16 bfloat;
typedef __attribute__((ext_vector_type(16))) __bf16 v16bf;
typedef __attribute__((ext_vector_type(8)))  float  v8f;
typedef __attribute__((ext_vector_type(2)))  __bf16 bf16x2;

__device__ __forceinline__ v8f wmma_bf16(v16bf a, v16bf b, v8f c) {
  // D = A(16x32 bf16) * B(32x16 bf16) + C(16x16 f32)
  return __builtin_amdgcn_wmma_f32_16x16x32_bf16(false, a, false, b, (short)0, c,
                                                 false, false);
}

// ---- A fragment: 16x32 bf16 tile, row-major source, row stride `ld` elems.
// Contiguous k order (not ISA table order): lanes0-15 M=lane, K = 0..15; lanes16-31 K = 16..31
__device__ __forceinline__ v16bf load_a_frag(const bfloat* base, int ld) {
  const int lane = threadIdx.x & 31;
  const int m  = lane & 15;
  const int kb = (lane < 16) ? 0 : 16;
  const bfloat* row = base + (size_t)m * ld + kb;
  v16bf a;
#pragma unroll
  for (int p = 0; p < 8; ++p) {
    const int k = 2 * p;                             // contiguous: 2p
    bf16x2 w = *(const bf16x2*)(row + k);
    a[2 * p]     = w.x;
    a[2 * p + 1] = w.y;
  }
  return a;
}

// ---- B fragment: 32x16 bf16 tile; element (k,n) stored at baseT[n*ld + k]
// Contiguous k order (not ISA table order): lanes0-15 N=lane, K=0..15 (2/VGPR); lanes16-31 K=16..31
__device__ __forceinline__ v16bf load_b_frag(const bfloat* baseT, int ld) {
  const int lane = threadIdx.x & 31;
  const int n  = lane & 15;
  const int kb = (lane < 16) ? 0 : 16;
  const bfloat* col = baseT + (size_t)n * ld + kb;
  v16bf b;
#pragma unroll
  for (int p = 0; p < 8; ++p) {
    bf16x2 w = *(const bf16x2*)(col + 2 * p);
    b[2 * p]     = w.x;
    b[2 * p + 1] = w.y;
  }
  return b;
}

__device__ __forceinline__ v8f vzero8() {
  v8f z = {0.f, 0.f, 0.f, 0.f, 0.f, 0.f, 0.f, 0.f};
  return z;
}

// Raw LDS byte offset of a __shared__ object: the LDS aperture maps flat
// addr[31:0] directly onto the LDS address (upper bits discarded).
__device__ __forceinline__ uint32_t lds_off(const void* p) {
  return (uint32_t)(uintptr_t)p;
}

// Async DMA: 16 bytes per lane, global -> LDS, tracked by ASYNCcnt.
__device__ __forceinline__ void async_to_lds_b128(uint32_t lds_byte, const void* g) {
  asm volatile("global_load_async_to_lds_b128 %0, %1, off"
               :: "v"(lds_byte), "v"(g) : "memory");
}

// ------------------------------------------------------------------ converts
__global__ void __launch_bounds__(256) cvt_bf16_kernel(const float* __restrict__ in,
                                                       bfloat* __restrict__ out, int n) {
  int i = blockIdx.x * 256 + threadIdx.x;
  if (i < n) out[i] = (bfloat)in[i];
}

// in[rows][cols] f32 -> out[cols][rows] bf16
__global__ void __launch_bounds__(256) transpose_bf16_kernel(const float* __restrict__ in,
                                                             bfloat* __restrict__ out,
                                                             int rows, int cols) {
  int i = blockIdx.x * 256 + threadIdx.x;
  if (i < rows * cols) {
    int r = i / cols, c = i % cols;
    out[(size_t)c * rows + r] = (bfloat)in[i];
  }
}

// --------------------------------------------------- QKV GEMM + bias + RoPE
// C[m,n] = xb[4096x768] @ W  (block tile 128x128; 8 waves of 32x64)
// Epilogue: bias, RoPE(q,k), q *= 1/8, scatter into Qh/Kh [h][s][d], Vt [h][d][s]
__global__ void __launch_bounds__(256) qkv_rope_kernel(
    const bfloat* __restrict__ xb, const bfloat* __restrict__ WqkvT,
    const float* __restrict__ b_qkv,
    bfloat* __restrict__ Qh, bfloat* __restrict__ Kh, bfloat* __restrict__ Vt) {
  const int tid = threadIdx.x, lane = tid & 31, wid = tid >> 5;
  const int wm = wid & 3, wn = wid >> 2;
  const int m0 = blockIdx.x * 128 + wm * 32;
  const int n0 = blockIdx.y * 128 + wn * 64;

  v8f acc[2][4];
#pragma unroll
  for (int i = 0; i < 2; ++i)
#pragma unroll
    for (int j = 0; j < 4; ++j) acc[i][j] = vzero8();

  for (int k0 = 0; k0 < DIM; k0 += 32) {
    v16bf a0 = load_a_frag(xb + (size_t)m0 * DIM + k0, DIM);
    v16bf a1 = load_a_frag(xb + (size_t)(m0 + 16) * DIM + k0, DIM);
#pragma unroll
    for (int j = 0; j < 4; ++j) {
      v16bf b = load_b_frag(WqkvT + (size_t)(n0 + j * 16) * DIM + k0, DIM);
      acc[0][j] = wmma_bf16(a0, b, acc[0][j]);
      acc[1][j] = wmma_bf16(a1, b, acc[1][j]);
    }
  }

  const int lm = lane & 15, hi = lane >> 4;
#pragma unroll
  for (int i = 0; i < 2; ++i) {
#pragma unroll
    for (int j = 0; j < 4; ++j) {
      const int n   = n0 + j * 16 + lm;
      const int sec = n / DIM;            // 0=Q 1=K 2=V (uniform across wave)
      const int h   = (n % DIM) / HD;
      const int jd  = n % HD;
      // emb[jd] = s * 10000^(-2*(jd%32)/64) = s * exp(-ln(1e4)/32 * (jd%32))
      const float invf = __expf(-0.28782313662425572f * (float)(jd & 31));
      const float bias = b_qkv[n];
#pragma unroll
      for (int r = 0; r < 8; ++r) {
        const int s = m0 + i * 16 + r + 8 * hi;     // sequence position
        float v = acc[i][j][r] + bias;
        float other = __shfl_xor(v, 1);             // RoPE partner (jd^1)
        if (sec == 2) {
          Vt[((size_t)h * HD + jd) * SEQ + s] = (bfloat)v;
        } else {
          float sv, cv;
          __sincosf((float)s * invf, &sv, &cv);
          const float rot = (jd & 1) ? other : -other;
          float o = v * cv + rot * sv;
          if (sec == 0) {
            o *= 0.125f;                            // 1/sqrt(hd) folded into Q
            Qh[((size_t)h * SEQ + s) * HD + jd] = (bfloat)o;
          } else {
            Kh[((size_t)h * SEQ + s) * HD + jd] = (bfloat)o;
          }
        }
      }
    }
  }
}

// -------------------------------------------------------- flash attention
// grid (SEQ/128, HEADS); 8 waves/block; each wave owns a 16-row Q strip.
// All waves share the same (head, kv) tiles, so K/V tiles are staged once per
// block into LDS via async global->LDS DMA, double-buffered to overlap the
// next tile's copy with the current tile's 8 WMMAs.
__global__ void __launch_bounds__(256) attn_kernel(
    const bfloat* __restrict__ Qh, const bfloat* __restrict__ Kh,
    const bfloat* __restrict__ Vt, bfloat* __restrict__ ctx) {
  // [buf][0]=K tile 32x64 (row=kv, col=d), [buf][1]=V tile 64x32 (row=d, col=kv)
  __shared__ bfloat KvBuf[2][2][2048];              // 16 KB
  __shared__ bfloat Ptile[8][16 * 32];              // 8 KB per-wave probs tiles

  const int tid = threadIdx.x, lane = tid & 31, wid = tid >> 5;
  const int lm = lane & 15, hi = lane >> 4;
  const int head = blockIdx.y;
  const int q0 = blockIdx.x * 128 + wid * 16;

  const bfloat* Qbase = Qh + ((size_t)head * SEQ + q0) * HD;
  const v16bf aq0 = load_a_frag(Qbase, HD);         // d = 0..31
  const v16bf aq1 = load_a_frag(Qbase + 32, HD);    // d = 32..63

  v8f accO[4];
  float mrow[8], lrow[8];
#pragma unroll
  for (int t = 0; t < 4; ++t) accO[t] = vzero8();
#pragma unroll
  for (int r = 0; r < 8; ++r) { mrow[r] = -1e30f; lrow[r] = 0.f; }

  // ---- stage helper: 256 threads x 16B cover each 4KB tile exactly once
  const int vr = tid >> 2, vq = tid & 3;            // V tile: row d, 16B quarter
  auto stage = [&](int buf, int kv) {
    // K tile: rows are consecutive s -> 4KB fully contiguous in Kh
    async_to_lds_b128(lds_off(&KvBuf[buf][0][tid * 8]),
                      Kh + ((size_t)head * SEQ + kv) * HD + tid * 8);
    // V tile: 64 rows of 64B, row stride SEQ elements in Vt
    async_to_lds_b128(lds_off(&KvBuf[buf][1][vr * 32 + vq * 8]),
                      Vt + ((size_t)head * HD + vr) * SEQ + kv + vq * 8);
  };

  int buf = 0;
  stage(0, 0);                                      // prologue: tile 0 in flight

  for (int kv0 = 0; kv0 < SEQ; kv0 += 32) {
    // issue next tile's DMA into the other buffer (wrap keeps code uniform)
    stage(buf ^ 1, (kv0 + 32) & (SEQ - 1));
    // wait for *current* tile (<=2 leaves only the 2 just-issued outstanding)
    asm volatile("s_wait_asynccnt 0x2" ::: "memory");
    __syncthreads();                                // publish tile to all waves

    const bfloat* Kt = &KvBuf[buf][0][0];
    const bfloat* Vb = &KvBuf[buf][1][0];

    // ---- S = Q @ K^T  (Q pre-scaled by 1/8)
    v16bf bk00 = load_b_frag(Kt, HD);               // kv tile 0, d 0..31
    v16bf bk01 = load_b_frag(Kt + 32, HD);          // kv tile 0, d 32..63
    v16bf bk10 = load_b_frag(Kt + 16 * HD, HD);     // kv tile 1
    v16bf bk11 = load_b_frag(Kt + 16 * HD + 32, HD);
    v8f s0 = vzero8(), s1 = vzero8();
    s0 = wmma_bf16(aq0, bk00, s0);
    s0 = wmma_bf16(aq1, bk01, s0);
    s1 = wmma_bf16(aq0, bk10, s1);
    s1 = wmma_bf16(aq1, bk11, s1);

    // ---- online softmax (each row's 16 cols live in one 16-lane half)
    bfloat* P = &Ptile[wid][0];
#pragma unroll
    for (int r = 0; r < 8; ++r) {
      float mx = fmaxf(s0[r], s1[r]);
#pragma unroll
      for (int o = 8; o >= 1; o >>= 1) mx = fmaxf(mx, __shfl_xor(mx, o));
      const float mnew = fmaxf(mrow[r], mx);
      const float corr = __expf(mrow[r] - mnew);
      const float p0 = __expf(s0[r] - mnew);
      const float p1 = __expf(s1[r] - mnew);
      float rs = p0 + p1;
#pragma unroll
      for (int o = 8; o >= 1; o >>= 1) rs += __shfl_xor(rs, o);
      lrow[r] = lrow[r] * corr + rs;
      mrow[r] = mnew;
#pragma unroll
      for (int t = 0; t < 4; ++t) accO[t][r] *= corr;
      P[(r + 8 * hi) * 32 + lm]      = (bfloat)p0;   // C-layout -> row-major LDS
      P[(r + 8 * hi) * 32 + 16 + lm] = (bfloat)p1;
    }
    __builtin_amdgcn_wave_barrier();

    // ---- O += P @ V   (P re-read from LDS in A-fragment layout)
    const v16bf ap = load_a_frag(P, 32);
#pragma unroll
    for (int t = 0; t < 4; ++t) {
      v16bf bv = load_b_frag(Vb + t * 16 * 32, 32);
      accO[t] = wmma_bf16(ap, bv, accO[t]);
    }

    __syncthreads();                 // everyone done reading buf before reuse
    buf ^= 1;
  }

  // ---- normalize, write ctx[s][h*64+d] (bf16 for the out-proj GEMM)
#pragma unroll
  for (int r = 0; r < 8; ++r) {
    const float inv = 1.f / lrow[r];
    const int srow = q0 + r + 8 * hi;
#pragma unroll
    for (int t = 0; t < 4; ++t)
      ctx[(size_t)srow * DIM + head * HD + t * 16 + lm] = (bfloat)(accO[t][r] * inv);
  }
}

// -------------------------------------------------------- output projection
__global__ void __launch_bounds__(256) out_proj_kernel(
    const bfloat* __restrict__ ctx, const bfloat* __restrict__ WoutT,
    const float* __restrict__ b_out, float* __restrict__ out) {
  const int tid = threadIdx.x, lane = tid & 31, wid = tid >> 5;
  const int wm = wid & 3, wn = wid >> 2;
  const int m0 = blockIdx.x * 128 + wm * 32;
  const int n0 = blockIdx.y * 128 + wn * 64;

  v8f acc[2][4];
#pragma unroll
  for (int i = 0; i < 2; ++i)
#pragma unroll
    for (int j = 0; j < 4; ++j) acc[i][j] = vzero8();

  for (int k0 = 0; k0 < DIM; k0 += 32) {
    v16bf a0 = load_a_frag(ctx + (size_t)m0 * DIM + k0, DIM);
    v16bf a1 = load_a_frag(ctx + (size_t)(m0 + 16) * DIM + k0, DIM);
#pragma unroll
    for (int j = 0; j < 4; ++j) {
      v16bf b = load_b_frag(WoutT + (size_t)(n0 + j * 16) * DIM + k0, DIM);
      acc[0][j] = wmma_bf16(a0, b, acc[0][j]);
      acc[1][j] = wmma_bf16(a1, b, acc[1][j]);
    }
  }

  const int lm = lane & 15, hi = lane >> 4;
#pragma unroll
  for (int i = 0; i < 2; ++i)
#pragma unroll
    for (int j = 0; j < 4; ++j) {
      const int n = n0 + j * 16 + lm;
      const float bias = b_out[n];
#pragma unroll
      for (int r = 0; r < 8; ++r) {
        const int m = m0 + i * 16 + r + 8 * hi;
        out[(size_t)m * DIM + n] = acc[i][j][r] + bias;
      }
    }
}

// ---------------------------------------------------------------- launcher
extern "C" void kernel_launch(void* const* d_in, const int* in_sizes, int n_in,
                              void* d_out, int out_size, void* d_ws, size_t ws_size,
                              hipStream_t stream) {
  (void)in_sizes; (void)n_in; (void)out_size; (void)ws_size;
  const float* x     = (const float*)d_in[0];
  const float* W_qkv = (const float*)d_in[1];
  const float* b_qkv = (const float*)d_in[2];
  const float* W_out = (const float*)d_in[3];
  const float* b_out = (const float*)d_in[4];

  size_t off = 0;
  auto take = [&](size_t elems) -> bfloat* {
    bfloat* p = (bfloat*)((char*)d_ws + off);
    off += (elems * sizeof(bfloat) + 255) & ~(size_t)255;
    return p;
  };
  bfloat* xb    = take((size_t)SEQ * DIM);      // x in bf16
  bfloat* WqkvT = take((size_t)QKVN * DIM);     // W_qkv^T bf16  [2304][768]
  bfloat* WoutT = take((size_t)DIM * DIM);      // W_out^T bf16  [768][768]
  bfloat* Qh    = take((size_t)HEADS * SEQ * HD);  // [h][s][d], scaled+RoPE'd
  bfloat* Khd   = take((size_t)HEADS * SEQ * HD);  // [h][s][d], RoPE'd
  bfloat* Vt    = take((size_t)HEADS * SEQ * HD);  // [h][d][s]
  bfloat* ctxb  = take((size_t)SEQ * DIM);      // attention output bf16

  cvt_bf16_kernel<<<(SEQ * DIM + 255) / 256, 256, 0, stream>>>(x, xb, SEQ * DIM);
  transpose_bf16_kernel<<<(DIM * QKVN + 255) / 256, 256, 0, stream>>>(W_qkv, WqkvT, DIM, QKVN);
  transpose_bf16_kernel<<<(DIM * DIM + 255) / 256, 256, 0, stream>>>(W_out, WoutT, DIM, DIM);

  qkv_rope_kernel<<<dim3(SEQ / 128, QKVN / 128), 256, 0, stream>>>(xb, WqkvT, b_qkv,
                                                                   Qh, Khd, Vt);
  attn_kernel<<<dim3(SEQ / 128, HEADS), 256, 0, stream>>>(Qh, Khd, Vt, ctxb);
  out_proj_kernel<<<dim3(SEQ / 128, DIM / 128), 256, 0, stream>>>(ctxb, WoutT, b_out,
                                                                  (float*)d_out);
}
